// attention_knowledge_28810640622402
// MI455X (gfx1250) — matrix operand order repaired and checked by reading, not compiled, not run on hardware
//
#include <hip/hip_runtime.h>
#include <stdint.h>

#define Bsz 16
#define Tsz 256
#define KF  64
#define Dd  128
#define Aa  128
#define TILES_PER_WG 4
#define NTHREADS 128

typedef __attribute__((ext_vector_type(16))) __bf16 v16bf;
typedef __attribute__((ext_vector_type(8)))  float  v8f;
typedef __attribute__((ext_vector_type(16))) unsigned short us16;

__device__ __forceinline__ unsigned short f2bf(float f) {
    unsigned int u = __float_as_uint(f);
    u += 0x7FFFu + ((u >> 16) & 1u);     // round to nearest even
    return (unsigned short)(u >> 16);
}

// Hardware V_TANH_F32 on gfx1250 (TRANS op, co-executes with WMMA); branch-free fallback.
__device__ __forceinline__ float fast_tanh(float x) {
#if __has_builtin(__builtin_amdgcn_tanhf)
    return __builtin_amdgcn_tanhf(x);
#else
    // tanh(x) = 1 - 2/(1 + 2^(2*log2(e)*x)); saturates correctly at +-inf
    float e = __builtin_amdgcn_exp2f(x * 2.88539008177793f);
    return 1.0f - 2.0f * __builtin_amdgcn_rcpf(1.0f + e);
#endif
}

__device__ __forceinline__ float fast_exp(float x) {
    return __builtin_amdgcn_exp2f(x * 1.44269504088896f);   // v_exp_f32
}

__global__ __launch_bounds__(NTHREADS)
void fused_knowledge_attention(
    const float* __restrict__ know,          // [B,T,K,D]
    const float* __restrict__ token,         // [B,T,D]
    const float* __restrict__ topic,         // [B,D]
    const float* __restrict__ Wp,            // [D,A]
    const float* __restrict__ Wt,            // [D,A]
    const float* __restrict__ Wc,            // [D,A]
    const float* __restrict__ bias,          // [A]
    const float* __restrict__ wred,          // [A]
    const unsigned char* __restrict__ mask,  // [B,T,K] (bool)
    float* __restrict__ out)                 // [B,T,D]
{
    // B-operand (W_private) fragments: 32 frags (kk 0..3, nb 0..7) x 512 bf16
    __shared__ __align__(32) unsigned short Wbf[Dd * Aa];   // 32 KiB
    // A-operand (knowledge tile) fragments: 16 frags (mb 0..3, kk 0..3) x 512 bf16
    __shared__ __align__(32) unsigned short Abf[KF * Dd];   // 16 KiB
    __shared__ float addLds[Aa];
    __shared__ float wredLds[Aa];
    __shared__ float tokLds[Dd];
    __shared__ float topLds[Dd];
    __shared__ float vwLds[KF];
    __shared__ float eLds[KF];

    const int tid  = threadIdx.x;
    const int lane = tid & 31;
    const int wv   = tid >> 5;         // wave id 0..3 -> M-block

    // ---- once per WG: W_private -> bf16, swizzled into B-operand layout ----
    // B 32x16 layout used here (k as in A): lane = (n&15) + 16*((k>>4)&1); slot = k&15 (bf16 pairs along K)
    for (int i = 0; i < 32; ++i) {
        int p    = tid + NTHREADS * i;          // 0..4095 (4 K-rows each)
        int aCol = p & (Aa - 1);
        int kD0  = (p >> 7) << 2;               // 0,4,...,124
        float w0 = Wp[(kD0 + 0) * Aa + aCol];
        float w1 = Wp[(kD0 + 1) * Aa + aCol];
        float w2 = Wp[(kD0 + 2) * Aa + aCol];
        float w3 = Wp[(kD0 + 3) * Aa + aCol];
        int kk = kD0 >> 5;
        int nb = aCol >> 4;
        int ln = (aCol & 15) + (((kD0 >> 4) & 1) << 4);
        int off = ((kk * 8 + nb) << 9) + (ln << 4) + (kD0 & 15);
        uint32_t lo = (uint32_t)f2bf(w0) | ((uint32_t)f2bf(w1) << 16);
        uint32_t hi = (uint32_t)f2bf(w2) | ((uint32_t)f2bf(w3) << 16);
        *(uint2*)&Wbf[off] = make_uint2(lo, hi);
    }
    wredLds[tid] = wred[tid];

    for (int t = 0; t < TILES_PER_WG; ++t) {
        const int bt = blockIdx.x * TILES_PER_WG + t;
        const float* kt = know + (size_t)bt * (KF * Dd);

        // ---- knowledge tile -> bf16, swizzled into A-operand layout ----
        // A 16x32 bf16 layout used here (k as in B): half = (k>>4)&1; slot = k&15; lane = row + 16*half
        for (int j = 0; j < 16; ++j) {
            int f  = tid + NTHREADS * j;        // float4 index 0..2047
            int k  = f >> 5;                    // row 0..63
            int d0 = (f & 31) << 2;             // col 0,4,...,124
            const float4 v = *(const float4*)(kt + k * Dd + d0);
            int mb = k >> 4, lr = k & 15;
            int kk = d0 >> 5, r0 = d0 & 31;
            int hf   = (r0 >> 4) & 1;
            int slot = r0 & 15;
            int ln   = lr + (hf << 4);
            int off  = ((mb * 4 + kk) << 9) + (ln << 4) + slot;
            uint32_t lo = (uint32_t)f2bf(v.x) | ((uint32_t)f2bf(v.y) << 16);
            uint32_t hi = (uint32_t)f2bf(v.z) | ((uint32_t)f2bf(v.w) << 16);
            *(uint2*)&Abf[off] = make_uint2(lo, hi);
        }
        tokLds[tid] = token[(size_t)bt * Dd + tid];
        topLds[tid] = topic[(size_t)(bt >> 8) * Dd + tid];   // b = bt / T

        // prefetch next tile into cache while this tile computes
        if (t + 1 < TILES_PER_WG) {
            const float* nx = kt + KF * Dd;
            __builtin_prefetch(nx + tid * 32, 0, 1);          // 128B lines, first half
            __builtin_prefetch(nx + 4096 + tid * 32, 0, 1);   // second half
        }
        __syncthreads();

        // ---- additive vector: bias + token@W_token + topic@W_topic (per A-col) ----
        {
            float acc = bias[tid];
            #pragma unroll 4
            for (int d = 0; d < Dd; ++d) {
                acc = fmaf(tokLds[d], Wt[d * Aa + tid], acc);
                acc = fmaf(topLds[d], Wc[d * Aa + tid], acc);
            }
            addLds[tid] = acc;
        }
        __syncthreads();

        // ---- GEMM (bf16 WMMA, f32 acc) + tanh + reduce over A ----
        {
            float rowacc[8];
            #pragma unroll
            for (int r = 0; r < 8; ++r) rowacc[r] = 0.f;
            const us16* Ap = (const us16*)Abf;
            const us16* Bp = (const us16*)Wbf;
            const int colBase = lane & 15;
            #pragma unroll
            for (int nb = 0; nb < 8; ++nb) {
                v8f acc = {0.f, 0.f, 0.f, 0.f, 0.f, 0.f, 0.f, 0.f};
                #pragma unroll
                for (int kk = 0; kk < 4; ++kk) {
                    v16bf a = __builtin_bit_cast(v16bf, Ap[(wv * 4 + kk) * 32 + lane]);
                    v16bf b = __builtin_bit_cast(v16bf, Bp[(kk * 8 + nb) * 32 + lane]);
                    acc = __builtin_amdgcn_wmma_f32_16x16x32_bf16(
                        false, a, false, b, (short)0, acc, false, false);
                }
                const int col   = (nb << 4) + colBase;
                const float adv = addLds[col];
                const float wrv = wredLds[col];
                #pragma unroll
                for (int r = 0; r < 8; ++r) {
                    float tv = fast_tanh(acc[r] + adv);
                    rowacc[r] = fmaf(tv, wrv, rowacc[r]);
                }
            }
            // reduce the 16 columns of each row half-wave-wide
            #pragma unroll
            for (int r = 0; r < 8; ++r) {
                float s = rowacc[r];
                s += __shfl_xor(s, 1, 32);
                s += __shfl_xor(s, 2, 32);
                s += __shfl_xor(s, 4, 32);
                s += __shfl_xor(s, 8, 32);
                if ((lane & 15) == 0)
                    vwLds[(wv << 4) + r + ((lane >> 4) << 3)] = s;
            }
        }
        __syncthreads();

        // ---- masked exp (no max-subtraction, per reference) ----
        if (tid < KF) {
            float m = mask[(size_t)bt * KF + tid] ? 1.0f : 0.0f;
            eLds[tid] = fast_exp(vwLds[tid]) * m;
        }
        __syncthreads();

        // ---- normalize + weighted sum over K (f32, tile is L2-hot) ----
        {
            float S = 0.f;
            #pragma unroll 8
            for (int k = 0; k < KF; ++k) S += eLds[k];
            const float inv = 1.0f / (S + 1e-7f);
            float o = 0.f;
            #pragma unroll 4
            for (int k = 0; k < KF; ++k)
                o = fmaf(eLds[k], kt[k * Dd + tid], o);
            out[(size_t)bt * Dd + tid] = o * inv;
        }
        // next iteration's first write (Abf) is safe: eLds rewrites sit behind
        // the load-stage + add-stage + gemm-stage barriers of the next tile.
    }
}

extern "C" void kernel_launch(void* const* d_in, const int* in_sizes, int n_in,
                              void* d_out, int out_size, void* d_ws, size_t ws_size,
                              hipStream_t stream) {
    (void)in_sizes; (void)n_in; (void)out_size; (void)d_ws; (void)ws_size;
    const float* know  = (const float*)d_in[0];
    const float* token = (const float*)d_in[1];
    const float* topic = (const float*)d_in[2];
    const float* Wp    = (const float*)d_in[3];
    const float* Wt    = (const float*)d_in[4];
    const float* Wc    = (const float*)d_in[5];
    const float* bias  = (const float*)d_in[6];
    const float* wred  = (const float*)d_in[7];
    const unsigned char* mask = (const unsigned char*)d_in[8];
    float* out = (float*)d_out;

    dim3 grid((Bsz * Tsz) / TILES_PER_WG);   // 1024 workgroups
    fused_knowledge_attention<<<grid, NTHREADS, 0, stream>>>(
        know, token, topic, Wp, Wt, Wc, bias, wred, mask, out);
}
